// GroupedQueryAttention_68350109548805
// MI455X (gfx1250) — compile-verified, matrix operand order checked by reading, not run on hardware
//
#include <hip/hip_runtime.h>
#include <hip/hip_bf16.h>
#include <cstdint>

// ---------------------------------------------------------------------------
// GQA attention block for MI455X (gfx1250, wave32, WMMA).
// bf16 WMMA everywhere, fp32 accumulate. GEMMs use CDNA5 async global->LDS
// (ASYNCcnt) double-buffered staging; attention is flash-style per-wave.
// ---------------------------------------------------------------------------

typedef __bf16 bf16_t;
typedef __attribute__((ext_vector_type(16))) __bf16 v16bf;
typedef __attribute__((ext_vector_type(8)))  float  v8f;

#define B_    2
#define T_    2048
#define D_    2048
#define H_    16
#define G_    4
#define DH_   128
#define QKVN_ 3072   // D + 2*KV

union Frag16 { v16bf v; uint4 q[2]; };
union Frag32 { v8f  v; float f[8]; };

__device__ __forceinline__ v8f wmma_bf16(const v16bf& a, const v16bf& b, const v8f& c) {
  return __builtin_amdgcn_wmma_f32_16x16x32_bf16(false, a, false, b, (short)0, c, false, false);
}

// CDNA5 async global->LDS copy, 16B per lane, tracked by ASYNCcnt.
// lds_off: wave-relative LDS byte offset (low 32 bits of flat LDS address),
// gbase:   uniform 64-bit base (SGPR pair), voff: per-lane 32-bit byte offset.
__device__ __forceinline__ void async_b128(unsigned lds_off, unsigned long long gbase,
                                           unsigned voff) {
  asm volatile("global_load_async_to_lds_b128 %0, %1, %2 offset:0"
               :
               : "v"(lds_off), "v"(voff), "s"(gbase)
               : "memory");
}
__device__ __forceinline__ void wait_async_le4() {
  asm volatile("s_wait_asynccnt 4" ::: "memory");
}
__device__ __forceinline__ void wait_async_0() {
  asm volatile("s_wait_asynccnt 0" ::: "memory");
}

// ---------------------------------------------------------------------------
// Kernel 1a: fp32 -> bf16 linear conversion (for x)
// ---------------------------------------------------------------------------
__global__ __launch_bounds__(256) void cvt_f32_to_bf16(const float* __restrict__ src,
                                                       bf16_t* __restrict__ dst, int n4) {
  int i = blockIdx.x * 256 + threadIdx.x;
  if (i < n4) {
    float4 v = reinterpret_cast<const float4*>(src)[i];
    bf16_t o[4] = { (bf16_t)v.x, (bf16_t)v.y, (bf16_t)v.z, (bf16_t)v.w };
    reinterpret_cast<uint2*>(dst)[i] = *reinterpret_cast<uint2*>(o);
  }
}

// ---------------------------------------------------------------------------
// Kernel 1b: fp32 (K,N) -> bf16 (N,K) transpose+convert (for weights)
// ---------------------------------------------------------------------------
__global__ __launch_bounds__(256) void cvt_transpose_bf16(const float* __restrict__ src,
                                                          bf16_t* __restrict__ dst,
                                                          int Kdim, int Ndim) {
  __shared__ float tile[32][33];
  const int n0 = blockIdx.x * 32, k0 = blockIdx.y * 32;
  const int tx = threadIdx.x & 31, ty = threadIdx.x >> 5;   // 32 x 8
#pragma unroll
  for (int r = 0; r < 32; r += 8)
    tile[ty + r][tx] = src[(size_t)(k0 + ty + r) * Ndim + n0 + tx];
  __syncthreads();
#pragma unroll
  for (int r = 0; r < 32; r += 8)
    dst[(size_t)(n0 + ty + r) * Kdim + k0 + tx] = (bf16_t)tile[tx][ty + r];
}

// ---------------------------------------------------------------------------
// Kernel 2: bf16 GEMM  C(f32, MxN) = A(bf16, MxK rowmaj) * Bt(bf16, NxK rowmaj)^T
// Block tile 128x128, K-tile 32, 8 waves (4M x 2N), wave tile 32x64.
// Double-buffered LDS fed by global_load_async_to_lds_b128 (ASYNCcnt).
// ---------------------------------------------------------------------------
__global__ __launch_bounds__(256) void gemm_bf16(const bf16_t* __restrict__ A,
                                                 const bf16_t* __restrict__ Bt,
                                                 float* __restrict__ C,
                                                 int M, int N, int K) {
  __shared__ __align__(16) bf16_t As[2][128][32];
  __shared__ __align__(16) bf16_t Bs[2][128][32];

  const int tid  = threadIdx.x;
  const int lane = tid & 31;
  const int wave = tid >> 5;
  const int wm = wave >> 1, wn = wave & 1;
  const int hi = lane >> 4, ln = lane & 15, kb8 = hi * 8;
  const int bm = blockIdx.y * 128, bn = blockIdx.x * 128;

  const unsigned ldsA0 = (unsigned)(uintptr_t)&As[0][0][0];
  const unsigned ldsB0 = (unsigned)(uintptr_t)&Bs[0][0][0];
  const unsigned long long ga = (unsigned long long)A;
  const unsigned long long gb = (unsigned long long)Bt;

  // Each thread owns 2 x 16B chunks per matrix per stage (512 chunks / matrix).
  auto issue = [&](int stage, int k0) {
#pragma unroll
    for (int c = 0; c < 2; ++c) {
      int idx = tid + c * 256;          // 0..511
      int rr = idx >> 2, c8 = (idx & 3) * 8;
      unsigned loff = (unsigned)((stage * 128 * 32 + rr * 32 + c8) * 2);
      async_b128(ldsA0 + loff, ga, (unsigned)(((bm + rr) * K + k0 + c8) * 2));
      async_b128(ldsB0 + loff, gb, (unsigned)(((bn + rr) * K + k0 + c8) * 2));
    }
  };

  Frag32 acc[2][4];
#pragma unroll
  for (int i = 0; i < 2; ++i)
#pragma unroll
    for (int j = 0; j < 4; ++j)
#pragma unroll
      for (int e = 0; e < 8; ++e) acc[i][j].f[e] = 0.f;

  const int nt = K / 32;
  issue(0, 0);
  for (int kt = 0; kt < nt; ++kt) {
    const int cur = kt & 1;
    if (kt + 1 < nt) {
      issue(cur ^ 1, (kt + 1) * 32);    // overlap next-stage DMA with compute
      wait_async_le4();                 // older stage's 4 copies complete (in-order)
    } else {
      wait_async_0();
    }
    __syncthreads();

    Frag16 af[2];
#pragma unroll
    for (int mi = 0; mi < 2; ++mi) {
      int r = wm * 32 + mi * 16 + ln;
      af[mi].q[0] = *reinterpret_cast<const uint4*>(&As[cur][r][kb8]);
      af[mi].q[1] = *reinterpret_cast<const uint4*>(&As[cur][r][16 + kb8]);
    }
#pragma unroll
    for (int ni = 0; ni < 4; ++ni) {
      Frag16 bfg;
      int cidx = wn * 64 + ni * 16 + ln;
      bfg.q[0] = *reinterpret_cast<const uint4*>(&Bs[cur][cidx][kb8]);
      bfg.q[1] = *reinterpret_cast<const uint4*>(&Bs[cur][cidx][16 + kb8]);
#pragma unroll
      for (int mi = 0; mi < 2; ++mi)
        acc[mi][ni].v = wmma_bf16(af[mi].v, bfg.v, acc[mi][ni].v);
    }
    __syncthreads();
  }

#pragma unroll
  for (int mi = 0; mi < 2; ++mi)
#pragma unroll
    for (int ni = 0; ni < 4; ++ni) {
      int cc = bn + wn * 64 + ni * 16 + ln;
#pragma unroll
      for (int j = 0; j < 8; ++j) {
        int rr = bm + wm * 32 + mi * 16 + j + 8 * hi;
        C[(size_t)rr * N + cc] = acc[mi][ni].f[j];
      }
    }
}

// ---------------------------------------------------------------------------
// Kernel 3: split QKV, apply RoPE to Q/K, write bf16:
//   Q (B,H,T,Dh)   K (B,G,T,Dh)   Vt (B,G,Dh,T) (transposed for PV B-frags)
// ---------------------------------------------------------------------------
__global__ __launch_bounds__(256) void rope_split(const float* __restrict__ qkv,
                                                  const float* __restrict__ sinT,
                                                  const float* __restrict__ cosT,
                                                  bf16_t* __restrict__ Q,
                                                  bf16_t* __restrict__ Kd,
                                                  bf16_t* __restrict__ Vt) {
  const int t = blockIdx.x, b = blockIdx.y;
  const float* row = qkv + ((size_t)b * T_ + t) * QKVN_;
  for (int i = threadIdx.x; i < QKVN_; i += 256) {
    float val = row[i];
    if (i < D_) {
      int hh = i >> 7, d = i & 127;
      float o;
      if (d < 64) {
        float c = cosT[t * 64 + d], s = sinT[t * 64 + d];
        o = val * c - row[(hh << 7) + d + 64] * s;
      } else {
        float c = cosT[t * 64 + d - 64], s = sinT[t * 64 + d - 64];
        o = val * c + row[(hh << 7) + d - 64] * s;
      }
      Q[(((size_t)b * H_ + hh) * T_ + t) * DH_ + d] = (bf16_t)o;
    } else if (i < D_ + G_ * DH_) {
      int gg = (i - D_) >> 7, d = (i - D_) & 127;
      int base = D_ + (gg << 7);
      float o;
      if (d < 64) {
        float c = cosT[t * 64 + d], s = sinT[t * 64 + d];
        o = val * c - row[base + d + 64] * s;
      } else {
        float c = cosT[t * 64 + d - 64], s = sinT[t * 64 + d - 64];
        o = val * c + row[base + d - 64] * s;
      }
      Kd[(((size_t)b * G_ + gg) * T_ + t) * DH_ + d] = (bf16_t)o;
    } else {
      int gg = (i - D_ - G_ * DH_) >> 7, d = (i - D_ - G_ * DH_) & 127;
      Vt[(((size_t)b * G_ + gg) * DH_ + d) * T_ + t] = (bf16_t)val;
    }
  }
}

// ---------------------------------------------------------------------------
// Kernel 4: flash-style GQA attention (unchanged WMMA structure, + prefetch).
// ---------------------------------------------------------------------------
__global__ __launch_bounds__(256) void gqa_flash(const bf16_t* __restrict__ Q,
                                                 const bf16_t* __restrict__ Kd,
                                                 const bf16_t* __restrict__ Vt,
                                                 const int* __restrict__ doc,
                                                 bf16_t* __restrict__ O) {
  const int b = blockIdx.z, h = blockIdx.y, g = h >> 2;
  const int wave = threadIdx.x >> 5, lane = threadIdx.x & 31;
  const int hi = lane >> 4, ln = lane & 15, kb8 = hi * 8;
  const int qb = blockIdx.x * 128 + wave * 16;
  const size_t qoff = ((size_t)b * H_ + h) * T_ * DH_;
  const size_t koff = ((size_t)b * G_ + g) * T_ * DH_;
  const size_t voff = ((size_t)b * G_ + g) * DH_ * T_;
  __shared__ __align__(16) bf16_t Plds[8][16][32];

  Frag16 qf[4];
  const int qrow = qb + ln;
#pragma unroll
  for (int w = 0; w < 4; ++w) {
    qf[w].q[0] = *reinterpret_cast<const uint4*>(Q + qoff + (size_t)qrow * DH_ + w * 32 + kb8);
    qf[w].q[1] = *reinterpret_cast<const uint4*>(Q + qoff + (size_t)qrow * DH_ + w * 32 + 16 + kb8);
  }

  float m[8], l[8];
  Frag32 o[8];
#pragma unroll
  for (int j = 0; j < 8; ++j) { m[j] = -1e30f; l[j] = 0.f; }
#pragma unroll
  for (int f = 0; f < 8; ++f)
#pragma unroll
    for (int j = 0; j < 8; ++j) o[f].f[j] = 0.f;

  int docq[8];
#pragma unroll
  for (int j = 0; j < 8; ++j) docq[j] = doc[b * T_ + qb + j + 8 * hi];

  const float scale = 0.088388347648318447f;  // 1/sqrt(128)

  for (int kt = 0; kt < qb + 16; kt += 32) {
    Frag32 s0, s1;
#pragma unroll
    for (int j = 0; j < 8; ++j) { s0.f[j] = 0.f; s1.f[j] = 0.f; }
    const int key0 = kt + ln, key1 = kt + 16 + ln;
#pragma unroll
    for (int w = 0; w < 4; ++w) {
      Frag16 kf;
      kf.q[0] = *reinterpret_cast<const uint4*>(Kd + koff + (size_t)key0 * DH_ + w * 32 + kb8);
      kf.q[1] = *reinterpret_cast<const uint4*>(Kd + koff + (size_t)key0 * DH_ + w * 32 + 16 + kb8);
      s0.v = wmma_bf16(qf[w].v, kf.v, s0.v);
    }
#pragma unroll
    for (int w = 0; w < 4; ++w) {
      Frag16 kf;
      kf.q[0] = *reinterpret_cast<const uint4*>(Kd + koff + (size_t)key1 * DH_ + w * 32 + kb8);
      kf.q[1] = *reinterpret_cast<const uint4*>(Kd + koff + (size_t)key1 * DH_ + w * 32 + 16 + kb8);
      s1.v = wmma_bf16(qf[w].v, kf.v, s1.v);
    }
    if (kt + 32 < qb + 16) {   // prefetch next key tile (global_prefetch_b8)
      __builtin_prefetch(Kd + koff + (size_t)(key0 + 32) * DH_ + kb8, 0, 1);
      __builtin_prefetch(Vt + voff + (size_t)ln * T_ + kt + 32, 0, 1);
    }
    const int dk0 = doc[b * T_ + key0];
    const int dk1 = doc[b * T_ + key1];

    float alpha[8], p0[8], p1[8];
#pragma unroll
    for (int j = 0; j < 8; ++j) {
      const int row = qb + j + 8 * hi;
      const bool ok0 = (key0 <= row) && (dk0 == docq[j]);
      const bool ok1 = (key1 <= row) && (dk1 == docq[j]);
      float v0 = ok0 ? s0.f[j] * scale : -1e30f;
      float v1 = ok1 ? s1.f[j] * scale : -1e30f;
      float rm = fmaxf(v0, v1);
#pragma unroll
      for (int off = 8; off >= 1; off >>= 1)
        rm = fmaxf(rm, __shfl_xor(rm, off, 32));    // within 16-lane half
      const float mn = fmaxf(m[j], rm);
      alpha[j] = __expf(m[j] - mn);
      p0[j] = ok0 ? __expf(v0 - mn) : 0.f;
      p1[j] = ok1 ? __expf(v1 - mn) : 0.f;
      float rs = p0[j] + p1[j];
#pragma unroll
      for (int off = 8; off >= 1; off >>= 1)
        rs += __shfl_xor(rs, off, 32);
      l[j] = l[j] * alpha[j] + rs;
      m[j] = mn;
    }

#pragma unroll
    for (int j = 0; j < 8; ++j) {
      Plds[wave][j + 8 * hi][ln]      = (bf16_t)p0[j];
      Plds[wave][j + 8 * hi][16 + ln] = (bf16_t)p1[j];
    }
#pragma unroll
    for (int f = 0; f < 8; ++f)
#pragma unroll
      for (int j = 0; j < 8; ++j) o[f].f[j] *= alpha[j];

    Frag16 pf;
    pf.q[0] = *reinterpret_cast<const uint4*>(&Plds[wave][ln][kb8]);
    pf.q[1] = *reinterpret_cast<const uint4*>(&Plds[wave][ln][16 + kb8]);
#pragma unroll
    for (int f = 0; f < 8; ++f) {
      Frag16 vf;
      const int d = f * 16 + ln;
      vf.q[0] = *reinterpret_cast<const uint4*>(Vt + voff + (size_t)d * T_ + kt + kb8);
      vf.q[1] = *reinterpret_cast<const uint4*>(Vt + voff + (size_t)d * T_ + kt + 16 + kb8);
      o[f].v = wmma_bf16(pf.v, vf.v, o[f].v);
    }
  }

  float invl[8];
#pragma unroll
  for (int j = 0; j < 8; ++j) invl[j] = 1.0f / l[j];
#pragma unroll
  for (int f = 0; f < 8; ++f)
#pragma unroll
    for (int j = 0; j < 8; ++j) {
      const int row = qb + j + 8 * hi;
      O[((size_t)b * T_ + row) * D_ + h * DH_ + f * 16 + ln] = (bf16_t)(o[f].f[j] * invl[j]);
    }
}

// ---------------------------------------------------------------------------
// Host launcher
// ---------------------------------------------------------------------------
extern "C" void kernel_launch(void* const* d_in, const int* in_sizes, int n_in,
                              void* d_out, int out_size, void* d_ws, size_t ws_size,
                              hipStream_t stream) {
  const float* x    = (const float*)d_in[0];
  const float* sinT = (const float*)d_in[1];
  const float* cosT = (const float*)d_in[2];
  const int*   doc  = (const int*)  d_in[3];
  const float* Wqkv = (const float*)d_in[4];
  const float* Wo   = (const float*)d_in[5];
  float* out = (float*)d_out;

  char* ws = (char*)d_ws;
  size_t off = 0;
  auto carve = [&](size_t bytes) -> char* {
    char* p = ws + off;
    off = (off + bytes + 255) & ~(size_t)255;
    return p;
  };
  bf16_t* xb     = (bf16_t*)carve((size_t)B_ * T_ * D_ * 2);       // x bf16 (M,K)
  bf16_t* wqkvT  = (bf16_t*)carve((size_t)QKVN_ * D_ * 2);         // W_qkv^T (N,K)
  bf16_t* woT    = (bf16_t*)carve((size_t)D_ * D_ * 2);            // W_o^T   (N,K)
  float*  qkv    = (float*) carve((size_t)B_ * T_ * QKVN_ * 4);    // QKV fp32
  bf16_t* Qb     = (bf16_t*)carve((size_t)B_ * H_ * T_ * DH_ * 2);
  bf16_t* Kb     = (bf16_t*)carve((size_t)B_ * G_ * T_ * DH_ * 2);
  bf16_t* Vtb    = (bf16_t*)carve((size_t)B_ * G_ * DH_ * T_ * 2);
  bf16_t* attnb  = (bf16_t*)carve((size_t)B_ * T_ * D_ * 2);
  (void)ws_size; (void)in_sizes; (void)n_in; (void)out_size;

  // 1) downconvert x; transpose+convert weights to (N,K) bf16
  cvt_f32_to_bf16<<<(B_ * T_ * D_ / 4 + 255) / 256, 256, 0, stream>>>(x, xb, B_ * T_ * D_ / 4);
  cvt_transpose_bf16<<<dim3(QKVN_ / 32, D_ / 32), 256, 0, stream>>>(Wqkv, wqkvT, D_, QKVN_);
  cvt_transpose_bf16<<<dim3(D_ / 32, D_ / 32), 256, 0, stream>>>(Wo, woT, D_, D_);

  // 2) QKV projection (4096 x 3072 x 2048)
  gemm_bf16<<<dim3(QKVN_ / 128, B_ * T_ / 128), 256, 0, stream>>>(xb, wqkvT, qkv, B_ * T_, QKVN_, D_);

  // 3) RoPE + split + V transpose
  rope_split<<<dim3(T_, B_), 256, 0, stream>>>(qkv, sinT, cosT, Qb, Kb, Vtb);

  // 4) flash GQA attention
  gqa_flash<<<dim3(T_ / 128, H_, B_), 256, 0, stream>>>(Qb, Kb, Vtb, doc, attnb);

  // 5) output projection (4096 x 2048 x 2048) -> fp32
  gemm_bf16<<<dim3(D_ / 128, B_ * T_ / 128), 256, 0, stream>>>(attnb, woT, out, B_ * T_, D_, D_);
}
